// DTWAttention_82214263980297
// MI455X (gfx1250) — matrix operand order repaired and checked by reading, not compiled, not run on hardware
//
#include <hip/hip_runtime.h>

typedef __attribute__((ext_vector_type(16))) _Float16 v16h;
typedef __attribute__((ext_vector_type(8)))  _Float16 v8h;
typedef __attribute__((ext_vector_type(8)))  float    v8f;
typedef __attribute__((ext_vector_type(4)))  float    v4f;

#define B_    64
#define N_    512
#define M_    512
#define DD_   64
#define GAMMA_ 0.1f
#define INVG_  10.0f
#define BIGF   1e30f

// ---------------------------------------------------------------------------
// Kernel 1: convert x,y (f32) -> f16 rows, and compute per-row squared norms.
// One thread per row (b,n); 2*B*N = 65536 threads total.
// ---------------------------------------------------------------------------
__global__ __launch_bounds__(256) void prep_kernel(
    const float* __restrict__ x, const float* __restrict__ y,
    _Float16* __restrict__ xh, _Float16* __restrict__ yh,
    float* __restrict__ xnorm, float* __restrict__ ynorm) {
  int id    = blockIdx.x * 256 + threadIdx.x;   // 0 .. 65535
  int which = id >> 15;                         // 0 = x, 1 = y
  int r     = id & (B_ * N_ - 1);
  const float* src = (which ? y : x) + (size_t)r * DD_;
  _Float16*    dst = (which ? yh : xh) + (size_t)r * DD_;
  const v4f* s4 = (const v4f*)src;
  float s = 0.0f;
#pragma unroll
  for (int c = 0; c < 8; ++c) {
    v4f a = s4[2 * c];
    v4f b = s4[2 * c + 1];
    s += a[0]*a[0] + a[1]*a[1] + a[2]*a[2] + a[3]*a[3]
       + b[0]*b[0] + b[1]*b[1] + b[2]*b[2] + b[3]*b[3];
    v8h h;
    h[0] = (_Float16)a[0]; h[1] = (_Float16)a[1];
    h[2] = (_Float16)a[2]; h[3] = (_Float16)a[3];
    h[4] = (_Float16)b[0]; h[5] = (_Float16)b[1];
    h[6] = (_Float16)b[2]; h[7] = (_Float16)b[3];
    ((v8h*)dst)[c] = h;
  }
  (which ? ynorm : xnorm)[r] = s;
}

// ---------------------------------------------------------------------------
// Kernel 2: batched WMMA GEMM epilogue-fused distance matrix:
//   D[b][m][n] = xx[b][m] + yy[b][n] - 2 * sum_k xh[b][m][k]*yh[b][n][k]
// One wave32 per 16x16 output tile, K=64 as 2 x v_wmma_f32_16x16x32_f16.
// ---------------------------------------------------------------------------
__global__ __launch_bounds__(256) void gemm_dist_kernel(
    const _Float16* __restrict__ xh, const _Float16* __restrict__ yh,
    const float* __restrict__ xnorm, const float* __restrict__ ynorm,
    float* __restrict__ Dm) {
  int wave = (blockIdx.x * 256 + threadIdx.x) >> 5;  // 0 .. 65535
  int lane = threadIdx.x & 31;
  int tn = wave & 31;           // N-tile
  int tm = (wave >> 5) & 31;    // M-tile
  int b  = wave >> 10;          // batch
  int i0 = tm * 16, j0 = tn * 16;

  const _Float16* xb = xh + (size_t)b * N_ * DD_;
  const _Float16* yb = yh + (size_t)b * M_ * DD_;

  int lh = lane & 15;
  int hi = lane >> 4;           // lane group 0/1
  int arow = i0 + lh;           // A: row = M = lane&15
  int bcol = j0 + lh;           // B: col = N = lane&15
  int akb  = hi * 16;           // A K-sub-offset, contiguous k = hi*16 + s
  int bkb  = hi * 16;           // B K-sub-offset, contiguous k = hi*16 + s

  v8f acc = {};
#pragma unroll
  for (int k0 = 0; k0 < DD_; k0 += 32) {
    // A 16x32 f16: halfs {k0+akb..+7} and {k0+8+akb..+7}
    v8h a0 = *(const v8h*)(xb + (size_t)arow * DD_ + k0 + akb);
    v8h a1 = *(const v8h*)(xb + (size_t)arow * DD_ + k0 + 8 + akb);
    v16h A = __builtin_shufflevector(a0, a1, 0,1,2,3,4,5,6,7,8,9,10,11,12,13,14,15);
    // B 32x16 f16 (B(k,n) = y[n][k]): 16 contiguous halfs at k0+bkb
    v16h Bm = *(const v16h*)(yb + (size_t)bcol * DD_ + k0 + bkb);
    acc = __builtin_amdgcn_wmma_f32_16x16x32_f16(
        /*neg_a=*/false, A, /*neg_b=*/false, Bm,
        /*c_mod=*/(short)0, acc, /*reuse_a=*/false, /*reuse_b=*/false);
  }

  // Epilogue: C layout -> lane l, vgpr v : M = v + 8*hi, N = l&15
  float ycn = ynorm[b * M_ + bcol];
  int moff = hi * 8;
  const float* xr   = xnorm + b * N_ + i0 + moff;
  float*       Drow = Dm + ((size_t)b * N_ + i0 + moff) * M_ + j0 + lh;
#pragma unroll
  for (int v = 0; v < 8; ++v) {
    Drow[(size_t)v * M_] = xr[v] + ycn - 2.0f * acc[v];
  }
}

// ---------------------------------------------------------------------------
// Kernel 3: soft-DTW anti-diagonal DP (one workgroup per batch, 512 threads),
// rolling 3 diagonals in LDS, then fused output scaling out = x * exp(-dist).
// ---------------------------------------------------------------------------
__global__ __launch_bounds__(512) void dtw_kernel(
    const float* __restrict__ Dm, const float* __restrict__ x,
    float* __restrict__ out) {
  __shared__ float V[3][N_ + 8];
  int b = blockIdx.x;
  int t = threadIdx.x;           // 0..511

  for (int i = t; i <= N_; i += 512) {
    V[0][i] = (i == 0) ? 0.0f : BIGF;   // diagonal p=0 : R[0,0]=0
    V[1][i] = BIGF;                      // diagonal p=1 : boundaries
  }
  __syncthreads();

  const float* Db = Dm + (size_t)b * N_ * M_;
  int i = t + 1;                 // DP row index 1..512

  for (int p = 2; p <= N_ + M_; ++p) {
    int cur = p % 3;
    int r1  = (p + 2) % 3;       // diagonal p-1
    int r2  = (p + 1) % 3;       // diagonal p-2
    int j   = p - i;
    float val = BIGF;
    if (j >= 1 && j <= M_) {
      float a  = V[r1][i - 1];   // R[i-1, j]
      float c2 = V[r1][i];       // R[i,   j-1]
      float c3 = V[r2][i - 1];   // R[i-1, j-1]
      float d  = Db[(size_t)(i - 1) * M_ + (j - 1)];
      float m  = fminf(a, fminf(c2, c3));
      float s  = __expf((m - a) * INVG_) + __expf((m - c2) * INVG_) +
                 __expf((m - c3) * INVG_);
      val = d + m - GAMMA_ * __logf(s);
    }
    V[cur][i] = val;
    if (t == 0) V[cur][0] = BIGF;
    __syncthreads();
  }

  float dist = V[(N_ + M_) % 3][N_];
  float w = __expf(-dist);
  const float* xb = x   + (size_t)b * N_ * DD_;
  float*       ob = out + (size_t)b * N_ * DD_;
  for (int e = t; e < N_ * DD_; e += 512) ob[e] = xb[e] * w;
}

// ---------------------------------------------------------------------------
extern "C" void kernel_launch(void* const* d_in, const int* in_sizes, int n_in,
                              void* d_out, int out_size, void* d_ws, size_t ws_size,
                              hipStream_t stream) {
  const float* x = (const float*)d_in[0];  // (64,512,64) f32
  const float* y = (const float*)d_in[1];  // (64,512,64) f32
  float* out = (float*)d_out;              // (64,512,64) f32

  char* ws = (char*)d_ws;
  _Float16* xh = (_Float16*)(ws);                                  //  4 MiB
  _Float16* yh = (_Float16*)(ws + ((size_t)4 << 20));              //  4 MiB
  float* xn    = (float*)   (ws + ((size_t)8 << 20));              // 128 KiB
  float* yn    = (float*)   (ws + ((size_t)8 << 20) + (128 << 10));// 128 KiB
  float* Dm    = (float*)   (ws + ((size_t)8 << 20) + (256 << 10));// 64 MiB (L2-resident)

  // 1) f32 -> f16 + row norms : 65536 rows
  prep_kernel<<<256, 256, 0, stream>>>(x, y, xh, yh, xn, yn);
  // 2) batched WMMA distance GEMM : 64*32*32 = 65536 waves
  gemm_dist_kernel<<<8192, 256, 0, stream>>>(xh, yh, xn, yn, Dm);
  // 3) soft-DTW DP + output scaling : one workgroup per batch
  dtw_kernel<<<B_, 512, 0, stream>>>(Dm, x, out);
}
